// HeteroGraphormerLayerComplete_54949811585312
// MI455X (gfx1250) — matrix operand order repaired and checked by reading, not compiled, not run on hardware
//
#include <hip/hip_runtime.h>
#include <hip/hip_bf16.h>
#include <stdint.h>

#define C_DIM 128
#define H_DIM 4
#define HASH_BITS 19
#define HASH_N (1u << HASH_BITS)
#define HASH_MASK (HASH_N - 1u)
#define HEMPTY 0xFFFFFFFFu

typedef __bf16 bf16_t;
typedef __attribute__((ext_vector_type(16))) __bf16 v16bf;
typedef __attribute__((ext_vector_type(8)))  float   v8f;

// ---------- helpers ----------
__device__ inline bf16_t f2bf(float f) {
  unsigned u = __float_as_uint(f);
  unsigned r = (u + 0x7FFFu + ((u >> 16) & 1u)) >> 16;   // RNE truncate to bf16
  unsigned short h = (unsigned short)r;
  return __builtin_bit_cast(bf16_t, h);
}

__device__ inline unsigned hash_u32(unsigned x) {
  x ^= x >> 16; x *= 0x7feb352dU;
  x ^= x >> 15; x *= 0x846ca68bU;
  x ^= x >> 16;
  return x;
}

__device__ inline void htab_add(unsigned* keys, unsigned* vals, unsigned key) {
  unsigned slot = hash_u32(key) & HASH_MASK;
  for (;;) {
    unsigned prev = atomicCAS(&keys[slot], HEMPTY, key);
    if (prev == HEMPTY || prev == key) { atomicAdd(&vals[slot], 1u); return; }
    slot = (slot + 1) & HASH_MASK;
  }
}
__device__ inline unsigned htab_get(const unsigned* keys, const unsigned* vals, unsigned key) {
  unsigned slot = hash_u32(key) & HASH_MASK;
  for (;;) {
    unsigned k = keys[slot];
    if (k == key)   return vals[slot];
    if (k == HEMPTY) return 0u;
    slot = (slot + 1) & HASH_MASK;
  }
}
__device__ inline void hset_add(unsigned* keys, unsigned key) {
  unsigned slot = hash_u32(key) & HASH_MASK;
  for (;;) {
    unsigned prev = atomicCAS(&keys[slot], HEMPTY, key);
    if (prev == HEMPTY || prev == key) return;
    slot = (slot + 1) & HASH_MASK;
  }
}
__device__ inline int hset_has(const unsigned* keys, unsigned key) {
  unsigned slot = hash_u32(key) & HASH_MASK;
  for (;;) {
    unsigned k = keys[slot];
    if (k == key)   return 1;
    if (k == HEMPTY) return 0;
    slot = (slot + 1) & HASH_MASK;
  }
}

// order-preserving float <-> uint for atomicMax-based segment max
__device__ inline unsigned enc_f32(float f) {
  unsigned b = __float_as_uint(f);
  return (b & 0x80000000u) ? ~b : (b | 0x80000000u);
}
__device__ inline float dec_f32(unsigned e) {
  unsigned b = (e & 0x80000000u) ? (e ^ 0x80000000u) : ~e;
  return __uint_as_float(b);
}

// ---------- generic fills / conversions ----------
__global__ void fill_u32_kernel(unsigned* __restrict__ p, long n, unsigned v) {
  long i = (long)blockIdx.x * blockDim.x + threadIdx.x;
  long stride = (long)gridDim.x * blockDim.x;
  for (; i < n; i += stride) p[i] = v;
}

__global__ void cvt_bf16_kernel(const float* __restrict__ in, bf16_t* __restrict__ out, long n) {
  long i = (long)blockIdx.x * blockDim.x + threadIdx.x;
  long stride = (long)gridDim.x * blockDim.x;
  for (; i < n; i += stride) out[i] = f2bf(in[i]);
}

// ---------- WMMA GEMM: C[M,128] = A[M,128] @ W[128,128]^T (+ bias) ----------
// Workgroup = 8 waves. The 128x128 bf16 W (32 KB) is staged in LDS once per
// workgroup (8x less L2 traffic), then each wave computes a 16x128 output
// strip: A fragments for all 4 K-chunks are prefetched in one clause (global
// latency paid once), B fragments come from LDS via ds_load_b128 (low
// latency), and 32 independent WMMAs accumulate into 64 VGPRs.
__global__ __launch_bounds__(256)
void gemm_bf16_wmma(const bf16_t* __restrict__ A, const bf16_t* __restrict__ W,
                    const float* __restrict__ bias, float* __restrict__ Cout,
                    int Mtiles) {
  __shared__ __align__(16) bf16_t lds_w[C_DIM * C_DIM];

  // cooperative W -> LDS copy (2048 uint4 / 256 threads = 8 each)
  for (int i = threadIdx.x; i < (C_DIM * C_DIM) / 8; i += 256) {
    ((uint4*)lds_w)[i] = ((const uint4*)W)[i];
  }
  __syncthreads();

  const int wave = threadIdx.x >> 5;
  const int lane = threadIdx.x & 31;
  const int tm   = blockIdx.x * 8 + wave;
  if (tm >= Mtiles) return;              // wave-uniform exit (after barrier)

  const int l15 = lane & 15;
  const int hi  = lane >> 4;             // 0 or 1
  const long arow = (long)(tm * 16 + l15) * C_DIM;
  const int ka = hi * 8;                 // A: lanes>=16 carry K+8 block
  const int kb = hi * 8;                 // B: lanes>=16 carry K+8 block

  // prefetch all A fragments (4 K-chunks) in one shot
  v16bf a[4];
#pragma unroll
  for (int kc = 0; kc < 4; ++kc) {
    ((uint4*)&a[kc])[0] = *(const uint4*)(A + arow + kc * 32 + ka);
    ((uint4*)&a[kc])[1] = *(const uint4*)(A + arow + kc * 32 + 16 + ka);
  }

  v8f acc[8];
#pragma unroll
  for (int tn = 0; tn < 8; ++tn) {
    float bval = bias ? bias[tn * 16 + l15] : 0.0f;
#pragma unroll
    for (int i = 0; i < 8; ++i) acc[tn][i] = bval;
  }

#pragma unroll
  for (int kc = 0; kc < 4; ++kc) {
#pragma unroll
    for (int tn = 0; tn < 8; ++tn) {
      v16bf b;
      const bf16_t* wp = lds_w + (long)(tn * 16 + l15) * C_DIM + kc * 32 + kb;
      ((uint4*)&b)[0] = *(const uint4*)(wp);
      ((uint4*)&b)[1] = *(const uint4*)(wp + 16);
      acc[tn] = __builtin_amdgcn_wmma_f32_16x16x32_bf16(false, a[kc], false, b, (short)0,
                                                        acc[tn], false, false);
    }
  }

#pragma unroll
  for (int tn = 0; tn < 8; ++tn) {
    const int col = tn * 16 + l15;
#pragma unroll
    for (int i = 0; i < 8; ++i) {
      int row = tm * 16 + hi * 8 + i;    // C layout: VGPR i -> M=i (+8 for lanes>=16)
      Cout[(long)row * C_DIM + col] = acc[tn][i];
    }
  }
}

// ---------- edge pass A: hash insert + degree counts ----------
__global__ __launch_bounds__(256)
void edge_build_kernel(const int* __restrict__ src, const int* __restrict__ dst, int E_,
                       unsigned mulDst, unsigned mulRev,
                       unsigned* __restrict__ ck, unsigned* __restrict__ cv,
                       unsigned* __restrict__ rk_other,
                       float* __restrict__ deg_src, float* __restrict__ deg_dst) {
  int e = blockIdx.x * 256 + threadIdx.x;
  if (e >= E_) return;
  unsigned s = (unsigned)src[e], d = (unsigned)dst[e];
  htab_add(ck, cv, s * mulDst + d);      // pair multiplicity for this relation
  hset_add(rk_other, d * mulRev + s);    // reversed pair -> other relation's recip set
  atomicAdd(&deg_src[s], 1.0f);
  atomicAdd(&deg_dst[d], 1.0f);
}

// ---------- edge pass B: counts / reciprocity lookup ----------
__global__ __launch_bounds__(256)
void edge_lookup_kernel(const int* __restrict__ src, const int* __restrict__ dst, int E_,
                        unsigned mulDst,
                        const unsigned* __restrict__ ck, const unsigned* __restrict__ cv,
                        const unsigned* __restrict__ rk_self,
                        float* __restrict__ counts, float* __restrict__ recips) {
  int e = blockIdx.x * 256 + threadIdx.x;
  if (e >= E_) return;
  unsigned s = (unsigned)src[e], d = (unsigned)dst[e];
  unsigned pid = s * mulDst + d;
  counts[e] = (float)htab_get(ck, cv, pid) - 1.0f;
  recips[e] = hset_has(rk_self, pid) ? 1.0f : 0.0f;
}

// ---------- edge pass 1: scores + segment max ----------
__global__ __launch_bounds__(256)
void edge_score_kernel(const int* __restrict__ src, const int* __restrict__ dst, int E_,
                       const float* __restrict__ Q, const float* __restrict__ K,
                       const float* __restrict__ t_src, const float* __restrict__ t_dst,
                       const float* __restrict__ hb, const float* __restrict__ beta,
                       const float* __restrict__ tau_raw, const float* __restrict__ gamma,
                       const float* __restrict__ delta,
                       const float* __restrict__ counts, const float* __restrict__ recips,
                       float* __restrict__ scores, unsigned* __restrict__ segmax) {
  int e = blockIdx.x * 256 + threadIdx.x;
  if (e >= E_) return;
  int s = src[e], d = dst[e];
  float dt = fabsf(t_dst[d] - t_src[s]) + 1e-6f;
  float tr = tau_raw[0];
  float tau = fmaxf(tr, 0.0f) + log1pf(expf(-fabsf(tr))) + 1e-6f;   // softplus + eps
  float tterm = -log1pf(dt / tau);
  float lc = log1pf(counts[e]);
  float rc = recips[e];
  const float4* q4 = (const float4*)(Q + (long)d * C_DIM);
  const float4* k4 = (const float4*)(K + (long)s * C_DIM);
#pragma unroll
  for (int h = 0; h < H_DIM; ++h) {
    float acc = 0.0f;
#pragma unroll
    for (int i = 0; i < 8; ++i) {
      float4 qq = q4[h * 8 + i], kk = k4[h * 8 + i];
      acc += qq.x * kk.x + qq.y * kk.y + qq.z * kk.z + qq.w * kk.w;
    }
    float sc = acc * 0.17677669529663687f          // 1/sqrt(32)
             + hb[h] + tterm * beta[h] + lc * gamma[h] + rc * delta[h];
    scores[(long)e * H_DIM + h] = sc;
    atomicMax(&segmax[(long)d * H_DIM + h], enc_f32(sc));
  }
}

// ---------- edge pass 2: exp + segment sum ----------
__global__ __launch_bounds__(256)
void edge_exp_kernel(const int* __restrict__ dst, int E_,
                     const unsigned* __restrict__ segmax,
                     float* __restrict__ scores, float* __restrict__ segsum) {
  int e = blockIdx.x * 256 + threadIdx.x;
  if (e >= E_) return;
  int d = dst[e];
#pragma unroll
  for (int h = 0; h < H_DIM; ++h) {
    float m = dec_f32(segmax[(long)d * H_DIM + h]);
    float ex = expf(scores[(long)e * H_DIM + h] - m);
    scores[(long)e * H_DIM + h] = ex;
    atomicAdd(&segsum[(long)d * H_DIM + h], ex);
  }
}

// ---------- edge pass 3: weighted-V scatter ----------
__global__ __launch_bounds__(256)
void edge_agg_kernel(const int* __restrict__ src, const int* __restrict__ dst, int E_,
                     const float* __restrict__ V, const float* __restrict__ w_exp,
                     const float* __restrict__ segsum, float* __restrict__ agg) {
  long gid = (long)blockIdx.x * 256 + threadIdx.x;
  int e = (int)(gid >> 5);
  int lane = (int)(gid & 31);
  if (e >= E_) return;
  int s = src[e], d = dst[e];
  int c = lane * 4;
  int h = c >> 5;
  float w = w_exp[(long)e * H_DIM + h] / segsum[(long)d * H_DIM + h];
  float4 v4 = *(const float4*)(V + (long)s * C_DIM + c);
  float* ap = agg + (long)d * C_DIM + c;
  atomicAdd(ap + 0, v4.x * w);
  atomicAdd(ap + 1, v4.y * w);
  atomicAdd(ap + 2, v4.z * w);
  atomicAdd(ap + 3, v4.w * w);
}

// ---------- final: + indeg*bo + deg + residual, layernorm (1 wave / row) ----------
__global__ __launch_bounds__(256)
void final_ln_kernel(float* __restrict__ io, const float* __restrict__ x,
                     const float* __restrict__ bo, const float* __restrict__ indeg,
                     const float* __restrict__ degA, const float* __restrict__ degB,
                     int N_) {
  int row = blockIdx.x * 8 + (threadIdx.x >> 5);
  int lane = threadIdx.x & 31;
  if (row >= N_) return;
  float ind = indeg[row];
  float deg = degA[row] + degB[row];
  float4 v  = *(const float4*)(io + (long)row * C_DIM + lane * 4);
  float4 xb = *(const float4*)(x  + (long)row * C_DIM + lane * 4);
  float4 bb = *(const float4*)(bo + lane * 4);
  v.x += ind * bb.x + deg + xb.x;
  v.y += ind * bb.y + deg + xb.y;
  v.z += ind * bb.z + deg + xb.z;
  v.w += ind * bb.w + deg + xb.w;
  float s  = v.x + v.y + v.z + v.w;
  float sq = v.x * v.x + v.y * v.y + v.z * v.z + v.w * v.w;
  for (int o = 16; o > 0; o >>= 1) { s += __shfl_xor(s, o, 32); sq += __shfl_xor(sq, o, 32); }
  float mu  = s * (1.0f / 128.0f);
  float var = fmaxf(sq * (1.0f / 128.0f) - mu * mu, 0.0f);
  float inv = rsqrtf(var + 1e-5f);
  float4 o4 = { (v.x - mu) * inv, (v.y - mu) * inv, (v.z - mu) * inv, (v.w - mu) * inv };
  *(float4*)(io + (long)row * C_DIM + lane * 4) = o4;
}

// ---------- host ----------
extern "C" void kernel_launch(void* const* d_in, const int* in_sizes, int n_in,
                              void* d_out, int out_size, void* d_ws, size_t ws_size,
                              hipStream_t stream) {
  const float* x_user = (const float*)d_in[0];
  const float* x_item = (const float*)d_in[1];
  const float* t_user = (const float*)d_in[2];
  const float* t_item = (const float*)d_in[3];
  const int*   eui    = (const int*)d_in[4];
  const int*   eiu    = (const int*)d_in[5];
  const float* Wq = (const float*)d_in[6];  const float* bq = (const float*)d_in[7];
  const float* Wk = (const float*)d_in[8];  const float* bk = (const float*)d_in[9];
  const float* Wv = (const float*)d_in[10]; const float* bv = (const float*)d_in[11];
  const float* Wo = (const float*)d_in[12]; const float* bo = (const float*)d_in[13];
  const float* hb    = (const float*)d_in[14];
  const float* beta  = (const float*)d_in[15];
  const float* taur  = (const float*)d_in[16];
  const float* gamma = (const float*)d_in[17];
  const float* delta = (const float*)d_in[18];

  const int NU = in_sizes[0] / C_DIM;
  const int NI = in_sizes[1] / C_DIM;
  const int E  = in_sizes[4] / 2;
  const int* su = eui;       const int* du = eui + E;
  const int* si = eiu;       const int* di = eiu + E;

  float* out_user = (float*)d_out;
  float* out_item = (float*)d_out + (long)NU * C_DIM;

  size_t off = 0;
  auto alloc = [&](size_t bytes) -> void* {
    void* p = (char*)d_ws + off;
    off += (bytes + 255) & ~(size_t)255;
    return p;
  };

  // --- zero-initialized region (one fill) ---
  size_t z0 = off;
  unsigned* hv_ui  = (unsigned*)alloc(HASH_N * 4);
  unsigned* hv_iu  = (unsigned*)alloc(HASH_N * 4);
  unsigned* smax_i = (unsigned*)alloc((long)NI * H_DIM * 4);
  unsigned* smax_u = (unsigned*)alloc((long)NU * H_DIM * 4);
  float* ssum_i = (float*)alloc((long)NI * H_DIM * 4);
  float* ssum_u = (float*)alloc((long)NU * H_DIM * 4);
  float* agg_i  = (float*)alloc((long)NI * C_DIM * 4);
  float* agg_u  = (float*)alloc((long)NU * C_DIM * 4);
  float* cnt_su = (float*)alloc((long)NU * 4);
  float* cnt_di = (float*)alloc((long)NU * 4);
  float* cnt_du = (float*)alloc((long)NI * 4);
  float* cnt_si = (float*)alloc((long)NI * 4);
  size_t zbytes = off - z0;
  // --- HEMPTY-initialized region (one fill) ---
  size_t e0 = off;
  unsigned* hk_ui = (unsigned*)alloc(HASH_N * 4);
  unsigned* hk_iu = (unsigned*)alloc(HASH_N * 4);
  unsigned* hr_ui = (unsigned*)alloc(HASH_N * 4);   // rev set for recips_ui
  unsigned* hr_iu = (unsigned*)alloc(HASH_N * 4);   // rev set for recips_iu
  size_t ebytes = off - e0;
  // --- no-init region ---
  bf16_t* xu_bf = (bf16_t*)alloc((long)NU * C_DIM * 2);
  bf16_t* xi_bf = (bf16_t*)alloc((long)NI * C_DIM * 2);
  bf16_t* wq_bf = (bf16_t*)alloc(C_DIM * C_DIM * 2);
  bf16_t* wk_bf = (bf16_t*)alloc(C_DIM * C_DIM * 2);
  bf16_t* wv_bf = (bf16_t*)alloc(C_DIM * C_DIM * 2);
  bf16_t* wo_bf = (bf16_t*)alloc(C_DIM * C_DIM * 2);
  float* Qu = (float*)alloc((long)NU * C_DIM * 4);
  float* Ku = (float*)alloc((long)NU * C_DIM * 4);
  float* Vu = (float*)alloc((long)NU * C_DIM * 4);
  float* Qi = (float*)alloc((long)NI * C_DIM * 4);
  float* Ki = (float*)alloc((long)NI * C_DIM * 4);
  float* Vi = (float*)alloc((long)NI * C_DIM * 4);
  bf16_t* aggbf_u = (bf16_t*)alloc((long)NU * C_DIM * 2);
  bf16_t* aggbf_i = (bf16_t*)alloc((long)NI * C_DIM * 2);
  float* sc_ui = (float*)alloc((long)E * H_DIM * 4);
  float* sc_iu = (float*)alloc((long)E * H_DIM * 4);
  float* cnts_ui = (float*)alloc((long)E * 4);
  float* cnts_iu = (float*)alloc((long)E * 4);
  float* rec_ui  = (float*)alloc((long)E * 4);
  float* rec_iu  = (float*)alloc((long)E * 4);

  auto cdiv = [](long a, long b) { return (int)((a + b - 1) / b); };

  // init
  fill_u32_kernel<<<4096, 256, 0, stream>>>((unsigned*)((char*)d_ws + z0), (long)(zbytes / 4), 0u);
  fill_u32_kernel<<<2048, 256, 0, stream>>>((unsigned*)((char*)d_ws + e0), (long)(ebytes / 4), HEMPTY);

  // bf16 conversions
  cvt_bf16_kernel<<<2048, 256, 0, stream>>>(x_user, xu_bf, (long)NU * C_DIM);
  cvt_bf16_kernel<<<2048, 256, 0, stream>>>(x_item, xi_bf, (long)NI * C_DIM);
  cvt_bf16_kernel<<<64, 256, 0, stream>>>(Wq, wq_bf, C_DIM * C_DIM);
  cvt_bf16_kernel<<<64, 256, 0, stream>>>(Wk, wk_bf, C_DIM * C_DIM);
  cvt_bf16_kernel<<<64, 256, 0, stream>>>(Wv, wv_bf, C_DIM * C_DIM);
  cvt_bf16_kernel<<<64, 256, 0, stream>>>(Wo, wo_bf, C_DIM * C_DIM);

  // QKV projections (WMMA): 8 waves / workgroup, W staged in LDS
  const int mtU = NU / 16, mtI = NI / 16;
  const int gGU = cdiv(mtU, 8), gGI = cdiv(mtI, 8);
  gemm_bf16_wmma<<<gGU, 256, 0, stream>>>(xu_bf, wq_bf, bq, Qu, mtU);
  gemm_bf16_wmma<<<gGU, 256, 0, stream>>>(xu_bf, wk_bf, bk, Ku, mtU);
  gemm_bf16_wmma<<<gGU, 256, 0, stream>>>(xu_bf, wv_bf, bv, Vu, mtU);
  gemm_bf16_wmma<<<gGI, 256, 0, stream>>>(xi_bf, wq_bf, bq, Qi, mtI);
  gemm_bf16_wmma<<<gGI, 256, 0, stream>>>(xi_bf, wk_bf, bk, Ki, mtI);
  gemm_bf16_wmma<<<gGI, 256, 0, stream>>>(xi_bf, wv_bf, bv, Vi, mtI);

  // pair-multiplicity / reciprocity hash build + degrees
  int gE = cdiv(E, 256);
  edge_build_kernel<<<gE, 256, 0, stream>>>(su, du, E, (unsigned)NI, (unsigned)NU,
                                            hk_ui, hv_ui, hr_iu, cnt_su, cnt_du);
  edge_build_kernel<<<gE, 256, 0, stream>>>(si, di, E, (unsigned)NU, (unsigned)NI,
                                            hk_iu, hv_iu, hr_ui, cnt_si, cnt_di);
  edge_lookup_kernel<<<gE, 256, 0, stream>>>(su, du, E, (unsigned)NI, hk_ui, hv_ui, hr_ui, cnts_ui, rec_ui);
  edge_lookup_kernel<<<gE, 256, 0, stream>>>(si, di, E, (unsigned)NU, hk_iu, hv_iu, hr_iu, cnts_iu, rec_iu);

  // attention: scores -> segment max -> exp/sum -> weighted scatter
  edge_score_kernel<<<gE, 256, 0, stream>>>(su, du, E, Qi, Ku, t_user, t_item,
                                            hb + 0, beta + 0, taur + 0, gamma + 0, delta + 0,
                                            cnts_ui, rec_ui, sc_ui, smax_i);
  edge_score_kernel<<<gE, 256, 0, stream>>>(si, di, E, Qu, Ki, t_item, t_user,
                                            hb + H_DIM, beta + H_DIM, taur + 1, gamma + H_DIM, delta + H_DIM,
                                            cnts_iu, rec_iu, sc_iu, smax_u);
  edge_exp_kernel<<<gE, 256, 0, stream>>>(du, E, smax_i, sc_ui, ssum_i);
  edge_exp_kernel<<<gE, 256, 0, stream>>>(di, E, smax_u, sc_iu, ssum_u);
  int gE32 = cdiv((long)E * 32, 256);
  edge_agg_kernel<<<gE32, 256, 0, stream>>>(su, du, E, Vu, sc_ui, ssum_i, agg_i);
  edge_agg_kernel<<<gE32, 256, 0, stream>>>(si, di, E, Vi, sc_iu, ssum_u, agg_u);

  // output projection hoisted after aggregation: lin = agg @ Wo^T  (bias folded as indeg*bo later)
  cvt_bf16_kernel<<<2048, 256, 0, stream>>>(agg_u, aggbf_u, (long)NU * C_DIM);
  cvt_bf16_kernel<<<2048, 256, 0, stream>>>(agg_i, aggbf_i, (long)NI * C_DIM);
  gemm_bf16_wmma<<<gGU, 256, 0, stream>>>(aggbf_u, wo_bf, nullptr, out_user, mtU);
  gemm_bf16_wmma<<<gGI, 256, 0, stream>>>(aggbf_i, wo_bf, nullptr, out_item, mtI);

  // + indeg*bo + total-degree bias + residual, then layernorm (in-place on d_out)
  final_ln_kernel<<<cdiv(NU, 8), 256, 0, stream>>>(out_user, x_user, bo, cnt_di, cnt_su, cnt_di, NU);
  final_ln_kernel<<<cdiv(NI, 8), 256, 0, stream>>>(out_item, x_item, bo, cnt_du, cnt_du, cnt_si, NI);
}
